// AdvancedGATModel_41386304864863
// MI455X (gfx1250) — matrix operand order repaired and checked by reading, not compiled, not run on hardware
//
#include <hip/hip_runtime.h>

#define NN 100000
#define E0 800000
#define ET 900000   // E + N self loops
#define HIDW 128

typedef __attribute__((ext_vector_type(16))) _Float16 v16h;
typedef __attribute__((ext_vector_type(8)))  _Float16 v8h;
typedef __attribute__((ext_vector_type(8)))  float    v8f;
typedef __attribute__((ext_vector_type(4)))  unsigned u32x4;
typedef __attribute__((ext_vector_type(8)))  int      i32x8;
typedef __attribute__((ext_vector_type(4)))  int      i32x4;

#if defined(__has_builtin)
#if __has_builtin(__builtin_amdgcn_tensor_load_to_lds) && \
    __has_builtin(__builtin_amdgcn_s_wait_tensorcnt)
#define USE_TDM 1
#endif
#endif

// ---------- helpers ----------
__device__ __forceinline__ unsigned enc_f(float f) {
  unsigned b = __float_as_uint(f);
  return (b & 0x80000000u) ? ~b : (b | 0x80000000u);
}
__device__ __forceinline__ float dec_f(unsigned e) {
  unsigned b = (e & 0x80000000u) ? (e & 0x7FFFFFFFu) : ~e;
  return __uint_as_float(b);
}

#ifdef USE_TDM
// Tensor DMA: 2-D f16 tile (rows x dim0 elems, row stride stride0 elems) -> LDS.
// D# per CDNA5 ISA ch.8: group0 {count,lds,global,type=2}, group1 {dims/strides};
// groups 2/3 zeroed (2-D tensor). 6-arg builtin variant (clang-23 / therock).
__device__ __forceinline__ void tdm_load_2d_f16(unsigned lds_off, const void* gaddr,
                                                unsigned dim0, unsigned rows,
                                                unsigned stride0) {
  unsigned long long ga = (unsigned long long)(uintptr_t)gaddr;
  u32x4 g0;
  g0[0] = 1u;                                   // count=1 (valid user descriptor)
  g0[1] = lds_off;                              // lds_addr (bytes)
  g0[2] = (unsigned)ga;                         // global_addr[31:0]
  g0[3] = (unsigned)((ga >> 32) & 0x01FFFFFFu) | (2u << 30);  // addr[56:32], type=2
  i32x8 g1;
  g1[0] = (int)(1u << 16);                      // wg_mask=0, data_size=1 (2 bytes)
  g1[1] = (int)((dim0 & 0xFFFFu) << 16);        // tensor_dim0[15:0]
  g1[2] = (int)((dim0 >> 16) | ((rows & 0xFFFFu) << 16));   // dim0 hi | tensor_dim1 lo
  g1[3] = (int)((rows >> 16) | ((dim0 & 0xFFFFu) << 16));   // dim1 hi | tile_dim0
  g1[4] = (int)(rows & 0xFFFFu);                // tile_dim1 (tile_dim2=0)
  g1[5] = (int)stride0;                         // tensor_dim0_stride[31:0]
  g1[6] = 0;                                    // stride0 hi | tensor_dim1_stride lo
  g1[7] = 0;
  i32x4 z4 = {};
  i32x8 z8 = {};
  __builtin_amdgcn_tensor_load_to_lds(g0, g1, z4, z4, z8, 0);
}
#endif

__global__ __launch_bounds__(256) void zero_u32_k(unsigned* __restrict__ p, long n) {
  long i = (long)blockIdx.x * 256 + threadIdx.x;
  if (i < n) p[i] = 0u;
}

__global__ __launch_bounds__(256) void cvt_f16_k(const float* __restrict__ in,
                                                 _Float16* __restrict__ out, long n) {
  long i = (long)blockIdx.x * 256 + threadIdx.x;
  if (i < n) out[i] = (_Float16)in[i];
}

// ---------- self-loop attr (PyG fill_value='mean') ----------
__global__ __launch_bounds__(256) void selfloop_stats_k(const long long* __restrict__ ei,
    const float* __restrict__ ea, float* __restrict__ deg, float* __restrict__ easum) {
  long e = (long)blockIdx.x * 256 + threadIdx.x;
  if (e >= E0) return;
  int dst = (int)ei[E0 + e];
  atomicAdd(&deg[dst], 1.0f);
  atomicAdd(&easum[dst], ea[e]);
}

__global__ __launch_bounds__(256) void loop_attr_k(const float* __restrict__ deg,
    const float* __restrict__ easum, float* __restrict__ loopa) {
  int n = blockIdx.x * 256 + threadIdx.x;
  if (n >= NN) return;
  loopa[n] = easum[n] / fmaxf(deg[n], 1.0f);
}

// ---------- WMMA GEMM: D[N x 128] = A[N x K] * B[K x 128] (+bias) ----------
// block = 256 threads = 8 waves; each wave owns one 16x16 output tile.
// A tile (16 x K) and the whole B panel (K x 128) staged into LDS via the
// Tensor Data Mover (wave 0 issues 2 DMAs, fences with s_wait_tensorcnt).
__global__ __launch_bounds__(256) void gemm_wmma_k(const _Float16* __restrict__ Af,
    const _Float16* __restrict__ Bf, const float* __restrict__ bias,
    float* __restrict__ D, int K) {
  __shared__ _Float16 As[16 * HIDW];
  __shared__ _Float16 Bs[HIDW * HIDW];
  const int row0  = blockIdx.x * 16;
  const int wave  = threadIdx.x >> 5;
  const int lane  = threadIdx.x & 31;
  const int nbase = wave * 16;

#ifdef USE_TDM
  if (wave == 0) {
    unsigned as_off = (unsigned)(uintptr_t)(void*)As;   // LDS aperture: low32 = offset
    unsigned bs_off = (unsigned)(uintptr_t)(void*)Bs;
    tdm_load_2d_f16(as_off, Af + (size_t)row0 * K, (unsigned)K, 16u, (unsigned)K);
    tdm_load_2d_f16(bs_off, Bf, (unsigned)HIDW, (unsigned)K, (unsigned)HIDW);
    __builtin_amdgcn_s_wait_tensorcnt(0);
  }
  __syncthreads();
#else
  const int kc = K >> 3;
  for (int t = threadIdx.x; t < 2 * K; t += 256) {      // A tile: 8-half chunks
    int r  = t / kc;
    int c8 = (t % kc) * 8;
    *(v8h*)(As + r * K + c8) = *(const v8h*)(Af + (size_t)(row0 + r) * K + c8);
  }
  for (int t = threadIdx.x; t < 16 * K; t += 256) {     // B panel: 8-half chunks
    int r  = t >> 4;
    int c8 = (t & 15) * 8;
    *(v8h*)(Bs + r * HIDW + c8) = *(const v8h*)(Bf + (size_t)r * HIDW + c8);
  }
  __syncthreads();
#endif

  v8f acc = {};
  const int m     = lane & 15;
  const int khalf = (lane >> 4) * 8;        // ISA A-layout: lanes 16-31 hold K+8 block
  for (int kb = 0; kb < K; kb += 32) {
    v8h alo = *(const v8h*)(As + m * K + kb + khalf);
    v8h ahi = *(const v8h*)(As + m * K + kb + 16 + khalf);
    v16h a  = __builtin_shufflevector(alo, ahi, 0,1,2,3,4,5,6,7,8,9,10,11,12,13,14,15);
    // B: lane holds col nbase + m, K rows in same order as A (khalf, 16 + khalf blocks)
    v16h b;
#pragma unroll
    for (int s = 0; s < 8; ++s) {
      b[s]     = Bs[(size_t)(kb + khalf + s) * HIDW + nbase + m];
      b[s + 8] = Bs[(size_t)(kb + 16 + khalf + s) * HIDW + nbase + m];
    }
    acc = __builtin_amdgcn_wmma_f32_16x16x32_f16(false, a, false, b,
                                                 (short)0, acc, false, false);
  }

  const int col = nbase + m;
  const int mhi = (lane >> 4) << 3;         // lanes 16-31 -> rows M+8
#pragma unroll
  for (int i = 0; i < 8; ++i) {
    float v = acc[i];
    if (bias) v += bias[col];
    D[(size_t)(row0 + i + mhi) * HIDW + col] = v;
  }
}

// ---------- per-node attention projections ----------
__global__ __launch_bounds__(256) void compute_al_k(const float* __restrict__ xh,
    const float* __restrict__ a_s, const float* __restrict__ a_d,
    float* __restrict__ al_s, float* __restrict__ al_d) {
  int i = blockIdx.x * 256 + threadIdx.x;       // i over N*H
  if (i >= NN * 4) return;
  int n = i >> 2, h = i & 3;
  const float* xr = xh + (size_t)n * HIDW + h * 32;
  const float* as = a_s + h * 32;
  const float* ad = a_d + h * 32;
  float s = 0.f, d = 0.f;
#pragma unroll
  for (int c = 0; c < 32; ++c) { s = fmaf(xr[c], as[c], s); d = fmaf(xr[c], ad[c], d); }
  al_s[i] = s; al_d[i] = d;
}

// we[h] = sum_c We[0, h*32+c] * a_edge[h, c]  (edge_attr is scalar per edge)
__global__ void compute_we_k(const float* __restrict__ We,
                             const float* __restrict__ a_e, float* __restrict__ we) {
  __shared__ float sh[128];
  int t = threadIdx.x;
  sh[t] = We[t] * a_e[t];
  __syncthreads();
  if (t < 4) {
    float s = 0.f;
#pragma unroll
    for (int c = 0; c < 32; ++c) s += sh[t * 32 + c];
    we[t] = s;
  }
}

// ---------- edge pass 1: logits + segment max ----------
__global__ __launch_bounds__(256) void edge_logits_max_k(const long long* __restrict__ ei,
    const float* __restrict__ ea, const float* __restrict__ loopa,
    const float* __restrict__ al_s, const float* __restrict__ al_d,
    const float* __restrict__ we, float* __restrict__ logits,
    unsigned* __restrict__ mbits) {
  long e = (long)blockIdx.x * 256 + threadIdx.x;
  if (e >= (long)ET) return;
  int src, dst; float a;
  if (e < E0) { src = (int)ei[e]; dst = (int)ei[E0 + e]; a = ea[e]; }
  else        { src = dst = (int)(e - E0); a = loopa[src]; }
#pragma unroll
  for (int h = 0; h < 4; ++h) {
    float lg = al_s[src * 4 + h] + al_d[dst * 4 + h] + a * we[h];
    lg = (lg > 0.f) ? lg : 0.2f * lg;                     // leaky_relu
    logits[e * 4 + h] = lg;
    atomicMax(&mbits[dst * 4 + h], enc_f(lg));
  }
}

// ---------- edge pass 2: exp, denominator, weighted scatter ----------
// one edge per wave32: 128 features -> 4 atomics/lane; head index == chunk index.
__global__ __launch_bounds__(256) void edge_aggregate_k(const long long* __restrict__ ei,
    const float* __restrict__ logits, const unsigned* __restrict__ mbits,
    const float* __restrict__ xh, float* __restrict__ den, float* __restrict__ g) {
  long e = (long)blockIdx.x * 8 + (threadIdx.x >> 5);
  int lane = threadIdx.x & 31;
  if (e >= (long)ET) return;
  int src, dst;
  if (e < E0) { src = (int)ei[e]; dst = (int)ei[E0 + e]; }
  else        { src = dst = (int)(e - E0); }
  const float* xr = xh + (size_t)src * HIDW;
  __builtin_prefetch(xr + lane * 4, 0, 0);                // global_prefetch_b8
  float ex[4];
#pragma unroll
  for (int h = 0; h < 4; ++h) {
    float m = dec_f(mbits[dst * 4 + h]);
    if (!(m > -1e30f && m < 1e30f)) m = 0.f;              // non-finite -> 0
    ex[h] = __expf(logits[e * 4 + h] - m);
  }
  if (lane < 4) atomicAdd(&den[dst * 4 + lane], ex[lane]);
  float* gr = g + (size_t)dst * HIDW;
#pragma unroll
  for (int r = 0; r < 4; ++r) {
    int j = lane + 32 * r;
    atomicAdd(&gr[j], ex[r] * xr[j]);
  }
}

// ---------- finalize: divide by den, add bias, accumulate LN stats (f64) ----------
__global__ __launch_bounds__(256) void finalize_reduce_k(float* __restrict__ g,
    const float* __restrict__ den, const float* __restrict__ bg,
    double* __restrict__ sums) {
  __shared__ double s1[256], s2[256];
  size_t i = (size_t)blockIdx.x * 256 + threadIdx.x;      // exact: N*128 % 256 == 0
  int j = (int)(i & 127); int n = (int)(i >> 7); int h = j >> 5;
  float v = g[i] / (den[n * 4 + h] + 1e-16f) + bg[j];
  g[i] = v;
  s1[threadIdx.x] = (double)v;
  s2[threadIdx.x] = (double)v * (double)v;
  __syncthreads();
  for (int s = 128; s > 0; s >>= 1) {
    if ((int)threadIdx.x < s) {
      s1[threadIdx.x] += s1[threadIdx.x + s];
      s2[threadIdx.x] += s2[threadIdx.x + s];
    }
    __syncthreads();
  }
  if (threadIdx.x == 0) {
    atomicAdd(&sums[0], s1[0]);
    atomicAdd(&sums[1], s2[0]);
  }
}

// ---------- graph layernorm + residual + relu ----------
__global__ __launch_bounds__(256) void norm_residual_k(const float* __restrict__ g,
    const double* __restrict__ sums, const float* __restrict__ lnw,
    const float* __restrict__ lnb, float* __restrict__ h) {
  size_t i = (size_t)blockIdx.x * 256 + threadIdx.x;
  if (i >= (size_t)NN * HIDW) return;
  int j = (int)(i & 127);
  const double cnt = (double)NN * (double)HIDW;
  float mean = (float)(sums[0] / cnt);
  float var  = (float)(sums[1] / cnt) - mean * mean;
  float inv  = rsqrtf(var + 1e-5f);
  float v = (g[i] - mean) * inv * lnw[j] + lnb[j] + h[i];
  h[i] = v > 0.f ? v : 0.f;
}

// ---------- output projection [N x 128] @ [128 x 5] ----------
__global__ __launch_bounds__(256) void out_proj_k(const float* __restrict__ h,
    const float* __restrict__ Wout, const float* __restrict__ bout,
    float* __restrict__ out) {
  int n = blockIdx.x * 256 + threadIdx.x;
  if (n >= NN) return;
  const float* hr = h + (size_t)n * HIDW;
  float acc[5];
#pragma unroll
  for (int o = 0; o < 5; ++o) acc[o] = bout[o];
  for (int j = 0; j < HIDW; ++j) {
    float v = hr[j];
#pragma unroll
    for (int o = 0; o < 5; ++o) acc[o] = fmaf(v, Wout[j * 5 + o], acc[o]);
  }
#pragma unroll
  for (int o = 0; o < 5; ++o) out[(size_t)n * 5 + o] = acc[o];
}

extern "C" void kernel_launch(void* const* d_in, const int* in_sizes, int n_in,
                              void* d_out, int out_size, void* d_ws, size_t ws_size,
                              hipStream_t stream) {
  const float*     x      = (const float*)d_in[0];
  const long long* ei     = (const long long*)d_in[1];   // int64 [2,E]
  const float*     ea     = (const float*)d_in[2];
  const float*     Win    = (const float*)d_in[3];
  const float*     b_in   = (const float*)d_in[4];
  const float*     Wg     = (const float*)d_in[5];
  const float*     bg     = (const float*)d_in[6];
  const float*     a_src  = (const float*)d_in[7];
  const float*     a_dst  = (const float*)d_in[8];
  const float*     We     = (const float*)d_in[9];
  const float*     a_edge = (const float*)d_in[10];
  const float*     ln_w   = (const float*)d_in[11];
  const float*     ln_b   = (const float*)d_in[12];
  const float*     Wout   = (const float*)d_in[13];
  const float*     bout   = (const float*)d_in[14];
  float*           out    = (float*)d_out;

  char* base = (char*)d_ws;
  size_t off = 0;
  auto carve = [&](size_t bytes) -> void* {
    void* p = base + off;
    off += (bytes + 255) & ~(size_t)255;
    return p;
  };
  float*    h      = (float*)carve(sizeof(float) * (size_t)NN * HIDW);
  float*    g      = (float*)carve(sizeof(float) * (size_t)NN * HIDW);
  float*    xh     = (float*)carve(sizeof(float) * (size_t)NN * HIDW);
  _Float16* hf     = (_Float16*)carve(sizeof(_Float16) * (size_t)NN * HIDW);
  _Float16* xf     = (_Float16*)carve(sizeof(_Float16) * (size_t)NN * 32);
  _Float16* Winf   = (_Float16*)carve(sizeof(_Float16) * 32 * HIDW);
  _Float16* Wgf    = (_Float16*)carve(sizeof(_Float16) * 3 * HIDW * HIDW);
  float*    al_s   = (float*)carve(sizeof(float) * (size_t)NN * 4);
  float*    al_d   = (float*)carve(sizeof(float) * (size_t)NN * 4);
  unsigned* mbits  = (unsigned*)carve(sizeof(unsigned) * (size_t)NN * 4);
  float*    den    = (float*)carve(sizeof(float) * (size_t)NN * 4);
  float*    logits = (float*)carve(sizeof(float) * (size_t)ET * 4);
  float*    deg    = (float*)carve(sizeof(float) * NN);
  float*    easum  = (float*)carve(sizeof(float) * NN);
  float*    loopa  = (float*)carve(sizeof(float) * NN);
  float*    wev    = (float*)carve(64);
  double*   sums   = (double*)carve(64);

  auto blk = [](long n) { return (unsigned)((n + 255) / 256); };

  // self-loop attrs
  zero_u32_k<<<blk(NN), 256, 0, stream>>>((unsigned*)deg, NN);
  zero_u32_k<<<blk(NN), 256, 0, stream>>>((unsigned*)easum, NN);
  selfloop_stats_k<<<blk(E0), 256, 0, stream>>>(ei, ea, deg, easum);
  loop_attr_k<<<blk(NN), 256, 0, stream>>>(deg, easum, loopa);

  // f16 conversions (one-time per call)
  cvt_f16_k<<<blk((long)NN * 32), 256, 0, stream>>>(x, xf, (long)NN * 32);
  cvt_f16_k<<<blk(32 * HIDW), 256, 0, stream>>>(Win, Winf, 32 * HIDW);
  cvt_f16_k<<<blk(3 * HIDW * HIDW), 256, 0, stream>>>(Wg, Wgf, 3 * HIDW * HIDW);

  // input projection: h = x @ Win + b_in  (WMMA, K=32)
  gemm_wmma_k<<<NN / 16, 256, 0, stream>>>(xf, Winf, b_in, h, 32);

  for (int l = 0; l < 3; ++l) {
    cvt_f16_k<<<blk((long)NN * HIDW), 256, 0, stream>>>(h, hf, (long)NN * HIDW);
    // xh = h @ Wg[l]  (WMMA, K=128)
    gemm_wmma_k<<<NN / 16, 256, 0, stream>>>(hf, Wgf + (size_t)l * HIDW * HIDW,
                                             nullptr, xh, HIDW);
    compute_al_k<<<blk((long)NN * 4), 256, 0, stream>>>(xh, a_src + l * 128,
                                                        a_dst + l * 128, al_s, al_d);
    compute_we_k<<<1, 128, 0, stream>>>(We + l * 128, a_edge + l * 128, wev);

    zero_u32_k<<<blk((long)NN * 4), 256, 0, stream>>>(mbits, (long)NN * 4);
    zero_u32_k<<<blk((long)NN * 4), 256, 0, stream>>>((unsigned*)den, (long)NN * 4);
    zero_u32_k<<<blk((long)NN * HIDW), 256, 0, stream>>>((unsigned*)g, (long)NN * HIDW);
    zero_u32_k<<<1, 4, 0, stream>>>((unsigned*)sums, 4);

    edge_logits_max_k<<<blk(ET), 256, 0, stream>>>(ei, ea, loopa, al_s, al_d,
                                                   wev, logits, mbits);
    edge_aggregate_k<<<(unsigned)((ET + 7) / 8), 256, 0, stream>>>(ei, logits, mbits,
                                                                   xh, den, g);
    finalize_reduce_k<<<(unsigned)(((long)NN * HIDW) / 256), 256, 0, stream>>>(
        g, den, bg + l * 128, sums);
    norm_residual_k<<<blk((long)NN * HIDW), 256, 0, stream>>>(g, sums, ln_w + l * 128,
                                                              ln_b + l * 128, h);
  }

  out_proj_k<<<blk(NN), 256, 0, stream>>>(h, Wout, bout, out);
}
